// CrossAttention_64991445123361
// MI455X (gfx1250) — matrix operand order repaired and checked by reading, not compiled, not run on hardware
//
#include <hip/hip_runtime.h>

// ---------------------------------------------------------------------------
// CDNA5 (gfx1250) cross-attention: bf16 WMMA, f32 accumulate.
// Double-buffered async global->LDS staging (ASYNCcnt prefetch depth 1),
// register-pipelined transposed-V staging, softmax scale folded into the
// Q projection, shuffle-parallel online softmax.
// ---------------------------------------------------------------------------

typedef __attribute__((ext_vector_type(16))) __bf16 v16bf;
typedef __attribute__((ext_vector_type(8)))  float  v8f;

#define DEV __device__ __forceinline__

// ---- constants for this problem -------------------------------------------
#define CA_B  2
#define CA_N  2048
#define CA_C  1024
#define CA_H  16
#define CA_D  64            // head dim
#define CA_BN (CA_B * CA_N) // 4096 rows
#define CA_SCALE 0.03125f   // 1024^-0.5 (full-dim scale per source)

// ---------------------------------------------------------------------------
DEV unsigned short f32_to_bf16(float f) {
  union { float f; unsigned u; } v; v.f = f;
  unsigned r = v.u + 0x7FFFu + ((v.u >> 16) & 1u);   // round-to-nearest-even
  return (unsigned short)(r >> 16);
}

union Frag16 { v16bf v; unsigned short u[16]; };

// LDS byte offset of a __shared__ object: flat-LDS addresses carry the
// wave-relative LDS offset in the low 32 bits (ISA 10.2 aperture mapping).
DEV unsigned lds_off(const void* p) {
  return (unsigned)(unsigned long long)p;
}

// Async DMA: 16 bytes per lane, global -> LDS, tracked by ASYNCcnt.
DEV void async_copy_b128(unsigned lds_byte_off, const void* gsrc) {
#if defined(__gfx1250__)
  asm volatile("global_load_async_to_lds_b128 %0, %1, off"
               :: "v"(lds_byte_off), "v"(gsrc)
               : "memory");
#endif
}

// Wait until ASYNCcnt <= N (async loads complete in order, so a non-zero
// threshold lets the next tile's DMA stay in flight).
template <int N>
DEV void async_wait() {
#if defined(__gfx1250__)
#if __has_builtin(__builtin_amdgcn_s_wait_asynccnt)
  __builtin_amdgcn_s_wait_asynccnt(N);
#else
  if constexpr (N == 0) asm volatile("s_wait_asynccnt 0x0" ::: "memory");
  else if constexpr (N == 2) asm volatile("s_wait_asynccnt 0x2" ::: "memory");
  else asm volatile("s_wait_asynccnt 0x4" ::: "memory");
#endif
#endif
}

// A fragment (16x32 bf16), src row-major with leading dim ld, origin (0,0)=s.
// ISA layout: lanes 0-15 hold M=lane, K=0..7 (v0-3) and K=16..23 (v4-7);
// lanes 16-31 hold K=8..15 and K=24..31.
DEV v16bf load_a_frag(const unsigned short* s, int ld, int lane) {
  const int m = lane & 15, h = lane >> 4;
  Frag16 r;
  const unsigned short* p = s + m * ld + h * 8;
#pragma unroll
  for (int j = 0; j < 8; ++j) r.u[j] = p[j];
  p = s + m * ld + 16 + h * 8;
#pragma unroll
  for (int j = 0; j < 8; ++j) r.u[8 + j] = p[j];
  return r.v;
}

// B fragment (32x16 bf16) where B[k,n] is stored as s[n*ld + k]
// (column n of B is a contiguous row of s). Lanes 0-15: K=0..7,16..23; lanes 16-31: K=8..15,24..31.
DEV v16bf load_b_frag(const unsigned short* s, int ld, int lane) {
  const int n = lane & 15, h = lane >> 4;
  Frag16 r;
  const unsigned short* p = s + n * ld + h * 8;
#pragma unroll
  for (int j = 0; j < 8; ++j) r.u[j] = p[j];
  p = s + n * ld + 16 + h * 8;
#pragma unroll
  for (int j = 0; j < 8; ++j) r.u[8 + j] = p[j];
  return r.v;
}

DEV v8f wmma_bf16(v16bf a, v16bf b, v8f c) {
  return __builtin_amdgcn_wmma_f32_16x16x32_bf16(false, a, false, b,
                                                 (short)0, c, false, false);
}

// ---------------------------------------------------------------------------
// fp32 -> bf16 elementwise conversion
// ---------------------------------------------------------------------------
__global__ __launch_bounds__(256)
void cvt_bf16_kernel(const float* __restrict__ src,
                     unsigned short* __restrict__ dst, int n) {
  int i = blockIdx.x * 256 + threadIdx.x;
  if (i < n) dst[i] = f32_to_bf16(src[i]);
}

// ---------------------------------------------------------------------------
// GEMM: Y[m,n] = alpha * (sum_k A[m,k] * W[n,k] + bias[n])
//   A: bf16 [M,K] row-major, W: bf16 [Nn,K] row-major (torch Linear weight),
//   out: bf16 [M,Nn] or f32 [M,Nn].
// Block tile 128x128, 8 waves (4x2), each wave 32x64 (2x4 tiles of 16x16).
// K staged in 32-deep chunks via double-buffered global_load_async_to_lds_b128:
// chunk c+1's DMA is in flight while chunk c feeds the WMMAs.
// Assumes M%128==0, Nn%128==0, K%32==0 (true for this problem).
// ---------------------------------------------------------------------------
template <bool F32OUT>
__global__ __launch_bounds__(256)
void gemm_wmma_kernel(const unsigned short* __restrict__ A,
                      const unsigned short* __restrict__ W,
                      const float* __restrict__ bias, float alpha,
                      void* __restrict__ out, int M, int Nn, int K) {
  __shared__ unsigned short As[2][128 * 32];   // 2 x 8 KB
  __shared__ unsigned short Ws[2][128 * 32];   // 2 x 8 KB

  const int t    = threadIdx.x;
  const int lane = t & 31;
  const int wid  = t >> 5;       // 0..7
  const int wm   = wid >> 1;     // 0..3 -> 32-row band
  const int wn   = wid & 1;      // 0..1 -> 64-col band
  const int m0   = blockIdx.y * 128;
  const int c0   = blockIdx.x * 128;

  v8f acc[2][4];
#pragma unroll
  for (int i = 0; i < 2; ++i)
#pragma unroll
    for (int j = 0; j < 4; ++j) acc[i][j] = (v8f)0.0f;

  const int ldrow = t >> 1;          // 0..127
  const int ldhal = t & 1;           // 16-element half of a 32-wide row

  const unsigned short* gArow = A + (size_t)(m0 + ldrow) * K + ldhal * 16;
  const unsigned short* gWrow = W + (size_t)(c0 + ldrow) * K + ldhal * 16;
  unsigned sAoff[2], sWoff[2];
#pragma unroll
  for (int bfi = 0; bfi < 2; ++bfi) {
    sAoff[bfi] = lds_off(&As[bfi][ldrow * 32 + ldhal * 16]);
    sWoff[bfi] = lds_off(&Ws[bfi][ldrow * 32 + ldhal * 16]);
  }

  auto issue_chunk = [&](int k0, int buf) {
    async_copy_b128(sAoff[buf],      gArow + k0);
    async_copy_b128(sAoff[buf] + 16, gArow + k0 + 8);
    async_copy_b128(sWoff[buf],      gWrow + k0);
    async_copy_b128(sWoff[buf] + 16, gWrow + k0 + 8);
  };

  const int nch = K >> 5;
  issue_chunk(0, 0);                       // prologue DMA

  for (int c = 0; c < nch; ++c) {
    const int kb = c & 1;
    if (c + 1 < nch) {
      issue_chunk((c + 1) << 5, kb ^ 1);   // WAR safe: prev iter's barrier
      async_wait<4>();                     // chunk c landed, c+1 in flight
    } else {
      async_wait<0>();
    }
    __syncthreads();

    v16bf af[2];
#pragma unroll
    for (int i = 0; i < 2; ++i)
      af[i] = load_a_frag(&As[kb][(wm * 32 + i * 16) * 32], 32, lane);
    v16bf bf_[4];
#pragma unroll
    for (int j = 0; j < 4; ++j)
      bf_[j] = load_b_frag(&Ws[kb][(wn * 64 + j * 16) * 32], 32, lane);

#pragma unroll
    for (int i = 0; i < 2; ++i)
#pragma unroll
      for (int j = 0; j < 4; ++j)
        acc[i][j] = wmma_bf16(af[i], bf_[j], acc[i][j]);

    __syncthreads();   // readers done before buffer kb is refilled at c+2
  }

  // store (C/D layout: lane n = lane&15, m = r + 8*(lane>>4))
  const int n = lane & 15, hh = lane >> 4;
#pragma unroll
  for (int i = 0; i < 2; ++i) {
#pragma unroll
    for (int j = 0; j < 4; ++j) {
      const int col = c0 + wn * 64 + j * 16 + n;
      const float bv = bias[col];
#pragma unroll
      for (int r = 0; r < 8; ++r) {
        const int row = m0 + wm * 32 + i * 16 + r + 8 * hh;
        const float v = (acc[i][j][r] + bv) * alpha;
        if (F32OUT) ((float*)out)[(size_t)row * Nn + col] = v;
        else ((unsigned short*)out)[(size_t)row * Nn + col] = f32_to_bf16(v);
      }
    }
  }
}

// ---------------------------------------------------------------------------
// Flash-style attention per (b, h, 64-row q tile).
// Q,K,V: bf16 [B,N,C] with head h at columns [h*64, h*64+64).
// Q is pre-scaled by 1/sqrt(DIM) (folded into its projection), so S needs
// no per-element multiply. K double-buffered through the async-DMA path;
// V pipelined through registers and stored transposed into LDS after the
// post-PV barrier. Online softmax runs 4 lanes per row with shuffle reduces.
// ---------------------------------------------------------------------------
__global__ __launch_bounds__(256)
void attn_wmma_kernel(const unsigned short* __restrict__ Q,
                      const unsigned short* __restrict__ K,
                      const unsigned short* __restrict__ V,
                      unsigned short* __restrict__ O) {
  __shared__ unsigned short Qs[64 * 64];      // 8 KB
  __shared__ unsigned short Ks[2][64 * 64];   // 2 x 8 KB
  __shared__ unsigned short Vt[64 * 64];      // 8 KB, transposed: Vt[d][kv]
  __shared__ float          Ss[64 * 64];      // 16 KB scores
  __shared__ unsigned short Ps[64 * 64];      // 8 KB probs (bf16)
  __shared__ float mrow[64], lrow[64], crow[64];

  const int t    = threadIdx.x;
  const int lane = t & 31;
  const int wid  = t >> 5;                  // 0..7
  const int b    = blockIdx.z;
  const int h    = blockIdx.y;
  const int q0   = blockIdx.x * 64;

  const size_t headoff = (size_t)h * CA_D;
  const unsigned short* Qg = Q + ((size_t)b * CA_N + q0) * CA_C + headoff;
  const unsigned short* Kg = K + (size_t)b * CA_N * CA_C + headoff;
  const unsigned short* Vg = V + (size_t)b * CA_N * CA_C + headoff;

  const int srow = t >> 2, spart = t & 3;   // 4 threads per 64-col row
  const unsigned short* gkRow = Kg + (size_t)srow * CA_C + spart * 16;
  const unsigned short* gvRow = Vg + (size_t)srow * CA_C + spart * 16;
  unsigned koff[2] = { lds_off(&Ks[0][srow * 64 + spart * 16]),
                       lds_off(&Ks[1][srow * 64 + spart * 16]) };

  // ---- prologue -----------------------------------------------------------
  {   // Q tile via async DMA
    const unsigned short* g = Qg + (size_t)srow * CA_C + spart * 16;
    const unsigned qoff = lds_off(Qs + srow * 64 + spart * 16);
    async_copy_b128(qoff,      g);
    async_copy_b128(qoff + 16, g + 8);
  }
  // K chunk 0 via async DMA
  async_copy_b128(koff[0],      gkRow);
  async_copy_b128(koff[0] + 16, gkRow + 8);
  // V chunk 0 through registers, stored transposed (no readers yet)
  {
    union { uint4 q[2]; unsigned short s[16]; } vrow;
    const uint4* gv = (const uint4*)gvRow;
    vrow.q[0] = gv[0]; vrow.q[1] = gv[1];
#pragma unroll
    for (int j = 0; j < 16; ++j)
      Vt[(spart * 16 + j) * 64 + srow] = vrow.s[j];
  }
  if (t < 64) { mrow[t] = -1e30f; lrow[t] = 0.0f; }

  v8f oacc[2];
  oacc[0] = (v8f)0.0f; oacc[1] = (v8f)0.0f;

  // ---- main loop over kv chunks ------------------------------------------
  for (int c = 0; c < CA_N / 64; ++c) {
    const int kb = c & 1;
    const bool have_next = (c + 1 < CA_N / 64);

    union { uint4 q[2]; unsigned short s[16]; } vnext;
    if (have_next) {
      // issue next K chunk DMA into the other buffer (WAR safe: barriers of
      // iteration c-1 separate it from that buffer's readers)
      const unsigned short* gk = gkRow + (size_t)(c + 1) * 64 * CA_C;
      async_copy_b128(koff[kb ^ 1],      gk);
      async_copy_b128(koff[kb ^ 1] + 16, gk + 8);
      // start next V chunk's global loads (registers -> no LDS hazard yet)
      const uint4* gv = (const uint4*)(gvRow + (size_t)(c + 1) * 64 * CA_C);
      vnext.q[0] = gv[0]; vnext.q[1] = gv[1];
      async_wait<2>();          // K(c) (and Q) landed; K(c+1) still in flight
    } else {
      async_wait<0>();
    }
    __syncthreads();            // B1: K(c), Vt(c) visible to all waves

    // S = (alpha*Q) . K^T : 4x4 tiles of 16x16, 2 per wave
#pragma unroll
    for (int i = 0; i < 2; ++i) {
      const int idx = wid * 2 + i;
      const int tm = idx >> 2, tn = idx & 3;
      v8f sc = (v8f)0.0f;
#pragma unroll
      for (int d = 0; d < 64; d += 32) {
        v16bf a = load_a_frag(Qs + tm * 16 * 64 + d, 64, lane);
        v16bf bb = load_b_frag(&Ks[kb][tn * 16 * 64 + d], 64, lane);
        sc = wmma_bf16(a, bb, sc);
      }
      const int n = lane & 15, hh = lane >> 4;
#pragma unroll
      for (int r = 0; r < 8; ++r)
        Ss[(tm * 16 + r + 8 * hh) * 64 + tn * 16 + n] = sc[r];
    }
    __syncthreads();            // B2

    // online softmax: 4 lanes per row, 16 columns each, shuffle reduces
    {
      const int r = t >> 2, p4 = t & 3;
      const float* srow_p = Ss + r * 64 + p4 * 16;
      const float mold = mrow[r];
      float mx = mold;
#pragma unroll
      for (int j = 0; j < 16; ++j) mx = fmaxf(mx, srow_p[j]);
      mx = fmaxf(mx, __shfl_xor(mx, 1));
      mx = fmaxf(mx, __shfl_xor(mx, 2));
      float l = 0.0f;
      unsigned short* prow_p = Ps + r * 64 + p4 * 16;
#pragma unroll
      for (int j = 0; j < 16; ++j) {
        const float p = __expf(srow_p[j] - mx);
        l += p;
        prow_p[j] = f32_to_bf16(p);
      }
      l += __shfl_xor(l, 1);
      l += __shfl_xor(l, 2);
      if (p4 == 0) {
        const float cc = __expf(mold - mx);
        mrow[r] = mx;
        lrow[r] = lrow[r] * cc + l;
        crow[r] = cc;
      }
    }
    __syncthreads();            // B3

    // O = O*c + P . V : 4x4 tiles, 2 per wave; Vt gives contiguous-K frags
#pragma unroll
    for (int i = 0; i < 2; ++i) {
      const int idx = wid * 2 + i;
      const int tm = idx >> 2, tn = idx & 3;
      const int hh = lane >> 4;
#pragma unroll
      for (int r = 0; r < 8; ++r)
        oacc[i][r] *= crow[tm * 16 + r + 8 * hh];
#pragma unroll
      for (int kk = 0; kk < 64; kk += 32) {
        v16bf a = load_a_frag(Ps + tm * 16 * 64 + kk, 64, lane);
        v16bf bb = load_b_frag(Vt + tn * 16 * 64 + kk, 64, lane);
        oacc[i] = wmma_bf16(a, bb, oacc[i]);
      }
    }
    __syncthreads();            // B4: PV readers of Vt done

    if (have_next) {            // commit next V chunk (transposed)
#pragma unroll
      for (int j = 0; j < 16; ++j)
        Vt[(spart * 16 + j) * 64 + srow] = vnext.s[j];
    }
  }

  // one reciprocal per row, then multiply in the epilogue
  if (t < 64) crow[t] = 1.0f / lrow[t];
  __syncthreads();

  // normalize and write O tile (bf16, [B,N,C] head-sliced layout)
  unsigned short* Og = O + ((size_t)b * CA_N + q0) * CA_C + headoff;
#pragma unroll
  for (int i = 0; i < 2; ++i) {
    const int idx = wid * 2 + i;
    const int tm = idx >> 2, tn = idx & 3;
    const int n = lane & 15, hh = lane >> 4;
#pragma unroll
    for (int r = 0; r < 8; ++r) {
      const int row = tm * 16 + r + 8 * hh;
      const float v = oacc[i][r] * crow[row];
      Og[(size_t)row * CA_C + tn * 16 + n] = f32_to_bf16(v);
    }
  }
}

// ---------------------------------------------------------------------------
extern "C" void kernel_launch(void* const* d_in, const int* in_sizes, int n_in,
                              void* d_out, int out_size, void* d_ws, size_t ws_size,
                              hipStream_t stream) {
  (void)in_sizes; (void)n_in; (void)out_size; (void)ws_size;

  const float* x  = (const float*)d_in[0];
  const float* qu = (const float*)d_in[1];
  const float* wq = (const float*)d_in[2];
  const float* bq = (const float*)d_in[3];
  const float* wk = (const float*)d_in[4];
  const float* bk = (const float*)d_in[5];
  const float* wv = (const float*)d_in[6];
  const float* bv = (const float*)d_in[7];
  const float* wo = (const float*)d_in[8];
  const float* bo = (const float*)d_in[9];

  const size_t nAct = (size_t)CA_BN * CA_C;   // 4096*1024
  const size_t nW   = (size_t)CA_C * CA_C;    // 1024*1024

  char* w = (char*)d_ws;
  auto take = [&](size_t bytes) -> char* {
    char* p = w;
    w += (bytes + 255) & ~(size_t)255;
    return p;
  };
  unsigned short* xb  = (unsigned short*)take(nAct * 2);
  unsigned short* qb  = (unsigned short*)take(nAct * 2);
  unsigned short* wqb = (unsigned short*)take(nW * 2);
  unsigned short* wkb = (unsigned short*)take(nW * 2);
  unsigned short* wvb = (unsigned short*)take(nW * 2);
  unsigned short* wob = (unsigned short*)take(nW * 2);
  unsigned short* Qp  = (unsigned short*)take(nAct * 2);
  unsigned short* Kp  = (unsigned short*)take(nAct * 2);
  unsigned short* Vp  = (unsigned short*)take(nAct * 2);
  unsigned short* AO  = (unsigned short*)take(nAct * 2);

  // 1) fp32 -> bf16
  cvt_bf16_kernel<<<(int)((nAct + 255) / 256), 256, 0, stream>>>(x,  xb, (int)nAct);
  cvt_bf16_kernel<<<(int)((nAct + 255) / 256), 256, 0, stream>>>(qu, qb, (int)nAct);
  cvt_bf16_kernel<<<(int)((nW + 255) / 256), 256, 0, stream>>>(wq, wqb, (int)nW);
  cvt_bf16_kernel<<<(int)((nW + 255) / 256), 256, 0, stream>>>(wk, wkb, (int)nW);
  cvt_bf16_kernel<<<(int)((nW + 255) / 256), 256, 0, stream>>>(wv, wvb, (int)nW);
  cvt_bf16_kernel<<<(int)((nW + 255) / 256), 256, 0, stream>>>(wo, wob, (int)nW);

  // 2) Q/K/V projections (bf16 out); softmax scale folded into Q
  dim3 gg(CA_C / 128, CA_BN / 128);   // (8, 32)
  gemm_wmma_kernel<false><<<gg, 256, 0, stream>>>(qb, wqb, bq, CA_SCALE, Qp, CA_BN, CA_C, CA_C);
  gemm_wmma_kernel<false><<<gg, 256, 0, stream>>>(xb, wkb, bk, 1.0f,     Kp, CA_BN, CA_C, CA_C);
  gemm_wmma_kernel<false><<<gg, 256, 0, stream>>>(xb, wvb, bv, 1.0f,     Vp, CA_BN, CA_C, CA_C);

  // 3) attention
  dim3 ga(CA_N / 64, CA_H, CA_B);     // (32, 16, 2)
  attn_wmma_kernel<<<ga, 256, 0, stream>>>(Qp, Kp, Vp, AO);

  // 4) output projection (f32 out -> d_out)
  gemm_wmma_kernel<true><<<gg, 256, 0, stream>>>(AO, wob, bo, 1.0f, d_out, CA_BN, CA_C, CA_C);
}
